// Model_17789754540447
// MI455X (gfx1250) — matrix operand order repaired and checked by reading, not compiled, not run on hardware
//
#include <hip/hip_runtime.h>

typedef __attribute__((ext_vector_type(16))) __bf16 v16bf;
typedef __attribute__((ext_vector_type(2)))  __bf16 v2bf;
typedef __attribute__((ext_vector_type(8)))  float  v8f;
typedef __attribute__((ext_vector_type(2)))  float  v2f;

#define Bb 4
#define Hh 16
#define Ss 1024
#define Dd 64

__device__ __forceinline__ unsigned cvt_pk_bf16(float a, float b) {
  v2f x = {a, b};
  union { v2bf v; unsigned u; } c;
  c.v = __builtin_convertvector(x, v2bf);   // v_cvt_pk_bf16_f32
  return c.u;
}
__device__ __forceinline__ unsigned short cvt_bf16(float a) {
  union { __bf16 b; unsigned short u; } c;
  c.b = (__bf16)a;
  return c.u;
}

union BF16Frag {
  v16bf v;
  unsigned short h[16];
  unsigned int u[8];
};

// ---------------------------------------------------------------------------
// Fused: wq = q @ W[h]  ->  qk = wq @ k^T (scaled)  ->  softmax over heads
//        -> dv = P @ V.   One block = (b, 16-row s-tile); 16 waves = 16 heads.
// K/V chunk loads are software-pipelined: chunk c+1's global_load_b128s issue
// right after chunk c's LDS store and are waited on a full compute-stage later.
// LDS: [0,4K) K bf16 [32][64] | [4K,8K) V^T bf16 [64][32]
//      [8K,40K) qk f32 [16][16][32] (aliases wq bf16 in prologue)
//      [40K,56K) P bf16 [16][16][32]
// ---------------------------------------------------------------------------
__global__ __launch_bounds__(512)
void attn_fused(const float* __restrict__ qg, const float* __restrict__ kg,
                const float* __restrict__ vg, const float* __restrict__ wg,
                const float* __restrict__ sfp, float* __restrict__ dv_ws) {
  __shared__ __align__(16) unsigned char smem[57344];
  unsigned short* key_bf = (unsigned short*)smem;            // [t][e]
  unsigned short* val_t  = (unsigned short*)(smem + 4096);   // [d][t]
  float*          qk_lds = (float*)(smem + 8192);            // [h][s][t] f32
  unsigned short* wq_sh  = (unsigned short*)(smem + 8192);   // prologue alias
  unsigned short* p_bf   = (unsigned short*)(smem + 40960);  // [h][s][t] bf16

  const int tid  = threadIdx.x;
  const int lane = tid & 31;
  const int h    = tid >> 5;          // wave id == head
  const int ln15 = lane & 15;
  const int hi   = lane >> 4;         // lane half (0/1)
  const int b    = blockIdx.x >> 6;   // S/16 = 64 s-tiles per batch
  const int s0   = (blockIdx.x & 63) << 4;
  const float sf = sfp[0];

  // ---- prologue: wq[16 x 64] = q[b, s0:s0+16, :] @ W[h]  (bf16 WMMA, f32 acc)
  BF16Frag qA[2];   // A-frags of query, K = d
  {
    const float* qrow = qg + ((long)b * Ss + s0 + ln15) * Dd;
    const int kb8 = hi ? 8 : 0;
#pragma unroll
    for (int ks = 0; ks < 2; ++ks)
#pragma unroll
      for (int i = 0; i < 4; ++i) {
        int k0 = ks * 32 + kb8 + 2 * i;       // slots 2i,2i+1 -> K k0,k0+1
        qA[ks].u[i]     = cvt_pk_bf16(qrow[k0], qrow[k0 + 1]);
        qA[ks].u[4 + i] = cvt_pk_bf16(qrow[k0 + 16], qrow[k0 + 17]);
      }
  }
  v8f wqC[4];
#pragma unroll
  for (int nt = 0; nt < 4; ++nt) {
    v8f acc = {};
#pragma unroll
    for (int ks = 0; ks < 2; ++ks) {
      BF16Frag wB;                            // B-frag of W[h]: (k=d, n=e)
      const int kb16 = hi ? 16 : 0;
#pragma unroll
      for (int j = 0; j < 16; ++j) {
        int kk = ks * 32 + (j < 8 ? (kb16 >> 1) + j : 16 + (kb16 >> 1) + (j - 8));
        wB.h[j] = cvt_bf16(wg[((long)h * Dd + kk) * Dd + nt * 16 + ln15]);
      }
      acc = __builtin_amdgcn_wmma_f32_16x16x32_bf16(false, qA[ks].v, false, wB.v,
                                                    (short)0, acc, false, false);
    }
    wqC[nt] = acc;
  }
  // park wq as bf16 row-major [s][e] in per-head LDS region
#pragma unroll
  for (int nt = 0; nt < 4; ++nt)
#pragma unroll
    for (int r = 0; r < 8; ++r)
      wq_sh[(h << 10) + (r + hi * 8) * 64 + nt * 16 + ln15] = cvt_bf16(wqC[nt][r]);
  __syncthreads();
  // reload wq as A-frags (K = e) for the qk matmuls
  BF16Frag wqA[2];
  {
    const unsigned short* base = wq_sh + (h << 10) + ln15 * 64;
    const int kb8 = hi ? 8 : 0;
#pragma unroll
    for (int ks = 0; ks < 2; ++ks)
#pragma unroll
      for (int i = 0; i < 4; ++i) {
        wqA[ks].u[i]     = *(const unsigned*)(base + ks * 32 + kb8 + 2 * i);
        wqA[ks].u[4 + i] = *(const unsigned*)(base + ks * 32 + kb8 + 16 + 2 * i);
      }
  }

  v8f dv[4] = {{}, {}, {}, {}};
  const float4* kchunks = (const float4*)(kg + (long)b * Ss * Dd);
  const float4* vchunks = (const float4*)(vg + (long)b * Ss * Dd);
  const int CHUNK4 = 32 * Dd / 4;           // float4s per chunk (=512 = blockDim)

  // pipeline prime: chunk 0 in registers
  float4 kv = kchunks[tid];
  float4 vv = vchunks[tid];

  for (int c = 0; c < Ss / 32; ++c) {
    __syncthreads();                        // prev-iter LDS readers done
    // stage held chunk into LDS as bf16 (K row-major, V transposed)
    {
      unsigned* dk = (unsigned*)key_bf + tid * 2;
      dk[0] = cvt_pk_bf16(kv.x, kv.y);
      dk[1] = cvt_pk_bf16(kv.z, kv.w);
      const int t  = (tid * 4) >> 6;
      const int d0 = (tid * 4) & 63;
      val_t[(d0 + 0) * 32 + t] = cvt_bf16(vv.x);
      val_t[(d0 + 1) * 32 + t] = cvt_bf16(vv.y);
      val_t[(d0 + 2) * 32 + t] = cvt_bf16(vv.z);
      val_t[(d0 + 3) * 32 + t] = cvt_bf16(vv.w);
    }
    // issue next chunk's global loads now; waited on next iteration,
    // hidden behind this chunk's WMMAs + softmax
    if (c + 1 < Ss / 32) {
      kv = kchunks[(c + 1) * CHUNK4 + tid];
      vv = vchunks[(c + 1) * CHUNK4 + tid];
    }
    __syncthreads();

    // qk tiles: M=16(s) x N=32(t), K=64(e)
#pragma unroll
    for (int tt = 0; tt < 2; ++tt) {
      v8f acc = {};
#pragma unroll
      for (int ks = 0; ks < 2; ++ks) {
        BF16Frag kB;                        // (k=e, n=t): pairs contiguous in e
        const unsigned short* kb = key_bf + (tt * 16 + ln15) * 64 + ks * 32 + (hi ? 8 : 0);
#pragma unroll
        for (int i = 0; i < 8; ++i) kB.u[i] = *(const unsigned*)(kb + 2 * i + (i >= 4 ? 8 : 0));
        acc = __builtin_amdgcn_wmma_f32_16x16x32_bf16(false, wqA[ks].v, false, kB.v,
                                                      (short)0, acc, false, false);
      }
#pragma unroll
      for (int r = 0; r < 8; ++r)
        qk_lds[(h * 16 + r + hi * 8) * 32 + tt * 16 + ln15] = acc[r] * sf;
    }
    __syncthreads();

    // softmax across the 16 heads: thread -> (s = tid>>5, t = tid&31)
    {
      const int s = tid >> 5, t = tid & 31;
      float m = -3.4e38f;
#pragma unroll
      for (int hh = 0; hh < 16; ++hh)
        m = fmaxf(m, qk_lds[(hh * 16 + s) * 32 + t]);
      float e[16], sum = 0.f;
#pragma unroll
      for (int hh = 0; hh < 16; ++hh) {
        e[hh] = __expf(qk_lds[(hh * 16 + s) * 32 + t] - m);
        sum += e[hh];
      }
      const float inv = 1.0f / sum;
#pragma unroll
      for (int hh = 0; hh < 16; ++hh)
        p_bf[(hh * 16 + s) * 32 + t] = cvt_bf16(e[hh] * inv);
    }
    __syncthreads();

    // dv += P @ V : M=16(s) x N=64(d), K=32(t)
    BF16Frag pA;
    {
      const unsigned short* pb = p_bf + (h * 16 + ln15) * 32;
      const int kb8 = hi ? 8 : 0;
#pragma unroll
      for (int i = 0; i < 4; ++i) {
        pA.u[i]     = *(const unsigned*)(pb + kb8 + 2 * i);
        pA.u[4 + i] = *(const unsigned*)(pb + kb8 + 16 + 2 * i);
      }
    }
#pragma unroll
    for (int dt = 0; dt < 4; ++dt) {
      BF16Frag vB;                          // (k=t, n=d): pairs contiguous in t
      const unsigned short* vb = val_t + (dt * 16 + ln15) * 32 + (hi ? 8 : 0);
#pragma unroll
      for (int i = 0; i < 8; ++i) vB.u[i] = *(const unsigned*)(vb + 2 * i + (i >= 4 ? 8 : 0));
      dv[dt] = __builtin_amdgcn_wmma_f32_16x16x32_bf16(false, pA.v, false, vB.v,
                                                       (short)0, dv[dt], false, false);
    }
  }

  // store dv [b,h,s,d] f32 (C/D layout: lane=N, VGPR r -> M = r / r+8)
  float* outp = dv_ws + (((long)b * Hh + h) * Ss + s0) * Dd;
#pragma unroll
  for (int dt = 0; dt < 4; ++dt)
#pragma unroll
    for (int r = 0; r < 8; ++r)
      outp[(r + hi * 8) * 64 + dt * 16 + ln15] = dv[dt][r];
}

// ---------------------------------------------------------------------------
// softmax over d (axis=3): one thread per (b,h,s) row of 64
// ---------------------------------------------------------------------------
__global__ __launch_bounds__(256)
void row_softmax_d(const float* __restrict__ dv_ws, float* __restrict__ sdv_ws) {
  const long row = (long)blockIdx.x * 256 + threadIdx.x;
  const float* p = dv_ws + row * Dd;
  float m = -3.4e38f;
  for (int i = 0; i < Dd; ++i) m = fmaxf(m, p[i]);
  float sum = 0.f;
  for (int i = 0; i < Dd; ++i) sum += __expf(p[i] - m);
  const float inv = 1.f / sum;
  float* o = sdv_ws + row * Dd;
  for (int i = 0; i < Dd; ++i) o[i] = __expf(p[i] - m) * inv;
}

// ---------------------------------------------------------------------------
// softmax over s (axis=2) of sdv, then out = result * dv.
// One block per (b,h); thread d owns one length-1024 column (L2 resident).
// ---------------------------------------------------------------------------
__global__ __launch_bounds__(64)
void col_softmax_s(const float* __restrict__ sdv_ws, const float* __restrict__ dv_ws,
                   float* __restrict__ out) {
  const long base = (long)blockIdx.x * Ss * Dd;
  const int d = threadIdx.x;
  float m = -3.4e38f;
  for (int s = 0; s < Ss; ++s) m = fmaxf(m, sdv_ws[base + (long)s * Dd + d]);
  float sum = 0.f;
  for (int s = 0; s < Ss; ++s) sum += __expf(sdv_ws[base + (long)s * Dd + d] - m);
  const float inv = 1.f / sum;
  for (int s = 0; s < Ss; ++s) {
    const long idx = base + (long)s * Dd + d;
    out[idx] = __expf(sdv_ws[idx] - m) * inv * dv_ws[idx];
  }
}

extern "C" void kernel_launch(void* const* d_in, const int* in_sizes, int n_in,
                              void* d_out, int out_size, void* d_ws, size_t ws_size,
                              hipStream_t stream) {
  const float* q  = (const float*)d_in[0];
  const float* k  = (const float*)d_in[1];
  const float* v  = (const float*)d_in[2];
  const float* w  = (const float*)d_in[3];
  const float* sf = (const float*)d_in[4];
  float* out = (float*)d_out;

  float* dv_ws  = (float*)d_ws;                               // 16.8 MB
  float* sdv_ws = dv_ws + (long)Bb * Hh * Ss * Dd;            // 16.8 MB

  attn_fused<<<Bb * (Ss / 16), 512, 0, stream>>>(q, k, v, w, sf, dv_ws);
  row_softmax_d<<<(Bb * Hh * Ss) / 256, 256, 0, stream>>>(dv_ws, sdv_ws);
  col_softmax_s<<<Bb * Hh, 64, 0, stream>>>(sdv_ws, dv_ws, out);
}
